// MultiHeadAttention_80204219285855
// MI455X (gfx1250) — matrix operand order repaired and checked by reading, not compiled, not run on hardware
//
#include <hip/hip_runtime.h>
#include <hip/hip_bf16.h>

// ---------------------------------------------------------------------------
// Types
// ---------------------------------------------------------------------------
typedef __attribute__((ext_vector_type(16))) __bf16 bf16x16;
typedef __attribute__((ext_vector_type(8)))  __bf16 bf16x8;
typedef __attribute__((ext_vector_type(4)))  __bf16 bf16x4;
typedef __attribute__((ext_vector_type(8)))  float  f32x8;

#define Bsz 4
#define Tsz 2048
#define Dsz 1024
#define Hsz 16
#define DHsz 64

// ---------------------------------------------------------------------------
// WMMA wrapper: D = A(16x32 bf16) * B(32x16 bf16) + C(16x16 f32)
// ---------------------------------------------------------------------------
static __device__ __forceinline__ f32x8 wmma_bf16(bf16x16 a, bf16x16 b, f32x8 c) {
    return __builtin_amdgcn_wmma_f32_16x16x32_bf16(
        /*neg_a=*/false, a, /*neg_b=*/false, b,
        /*c_mod=*/(short)0, c, /*reuse_a=*/false, /*reuse_b=*/false);
}

// A fragment (16x32 bf16, row-major source, row stride ld elems).
// Lane L: M = L%16 ; e0..7 -> K = hh*8+e, e8..15 -> K = 16+hh*8+(e-8)
static __device__ __forceinline__ bf16x16 frag_a_bf16(const __bf16* base, int ld, int lane) {
    int m = lane & 15, hh = lane >> 4;
    const __bf16* p = base + (size_t)m * ld + hh * 8;
    union { bf16x16 v; bf16x8 h[2]; } f;
    f.h[0] = *(const bf16x8*)(p);
    f.h[1] = *(const bf16x8*)(p + 16);
    return f.v;
}

// B fragment (32x16 K x N) from an N-major buffer: buf[n][k], row stride ld elems.
// Lane L: N = L%16 ; e0..7 -> K = hh*8+e, e8..15 -> K = 16+hh*8+(e-8)
static __device__ __forceinline__ bf16x16 frag_b_nmajor(const __bf16* base, int ld, int lane) {
    int n = lane & 15, hh = lane >> 4;
    const __bf16* p = base + (size_t)n * ld + hh * 8;
    union { bf16x16 v; bf16x8 h[2]; } f;
    f.h[0] = *(const bf16x8*)(p);
    f.h[1] = *(const bf16x8*)(p + 16);
    return f.v;
}

// ---------------------------------------------------------------------------
// Kernel 1: QKV GEMM  [8192 x 3072] = x[8192 x 1024] @ W_qkv + b_qkv
// Q,K written [B,H,T,DH] bf16; V written TRANSPOSED [B,H,DH,T] bf16 so the
// attention kernel's P@V B-fragments are contiguous global loads.
// Block = 128 threads (4 waves); block tile 64(M) x 64(N); wave tile 64 x 16.
// A tile staged once per block in LDS as bf16 (kills 4x redundant cvt+loads).
// ---------------------------------------------------------------------------
__global__ __launch_bounds__(128) void qkv_gemm_kernel(
    const float* __restrict__ X, const float* __restrict__ W,
    const float* __restrict__ bias,
    __bf16* __restrict__ Qo, __bf16* __restrict__ Ko, __bf16* __restrict__ Vo)
{
    __shared__ alignas(16) __bf16 At[64 * 40];   // [m][k], stride 40
    __shared__ alignas(16) __bf16 Bt[64 * 40];   // [n][k], stride 40
    const int lane = threadIdx.x & 31;
    const int wave = threadIdx.x >> 5;
    const int hh = lane >> 4, ln = lane & 15;
    const int n0 = blockIdx.x * 64;
    const int row0 = blockIdx.y * 64;

    f32x8 acc[4];
#pragma unroll
    for (int mi = 0; mi < 4; ++mi) acc[mi] = (f32x8){0, 0, 0, 0, 0, 0, 0, 0};

    for (int k0 = 0; k0 < Dsz; k0 += 32) {
        __syncthreads();
        // Stage A tile (64 m x 32 k) -> At[m][k] bf16 (contiguous b64 stores)
#pragma unroll
        for (int i = 0; i < 4; ++i) {
            int qi = threadIdx.x + i * 128;      // 0..511 quads
            int m  = qi >> 3;                    // 0..63
            int c4 = (qi & 7) * 4;               // 0..28
            float4 xv = *(const float4*)(X + (size_t)(row0 + m) * Dsz + k0 + c4);
            union { bf16x4 v; __bf16 e[4]; } cv;
            cv.e[0] = (__bf16)xv.x; cv.e[1] = (__bf16)xv.y;
            cv.e[2] = (__bf16)xv.z; cv.e[3] = (__bf16)xv.w;
            *(bf16x4*)(&At[m * 40 + c4]) = cv.v;
        }
        // Stage W tile (32 k x 64 n) transposed -> Bt[n][k]
#pragma unroll
        for (int i = 0; i < 4; ++i) {
            int qi = threadIdx.x + i * 128;
            int kk = qi >> 4;                    // 0..31
            int nq = (qi & 15) * 4;              // 0..60
            float4 wv = *(const float4*)(W + (size_t)(k0 + kk) * (3 * Dsz) + n0 + nq);
            Bt[(nq + 0) * 40 + kk] = (__bf16)wv.x;
            Bt[(nq + 1) * 40 + kk] = (__bf16)wv.y;
            Bt[(nq + 2) * 40 + kk] = (__bf16)wv.z;
            Bt[(nq + 3) * 40 + kk] = (__bf16)wv.w;
        }
        __syncthreads();

        bf16x16 bf = frag_b_nmajor(&Bt[wave * 16 * 40], 40, lane);
#pragma unroll
        for (int mi = 0; mi < 4; ++mi) {
            bf16x16 af = frag_a_bf16(&At[mi * 16 * 40], 40, lane);
            acc[mi] = wmma_bf16(af, bf, acc[mi]);
        }
    }

    // Epilogue: +bias; Q,K scatter [B,H,T,DH]; V vector-store [B,H,DH,T]
    const int nn = n0 + wave * 16 + ln;          // global output column (0..3071)
    const float bv = bias[nn];
    const int which = nn >> 10;                  // 0=Q 1=K 2=V (wave-uniform)
    const int rem = nn & 1023;
    const int head = rem >> 6;
    const int dh = rem & 63;
    if (which == 2) {
        // V transposed: rows r map to consecutive t -> one b128 store per acc
#pragma unroll
        for (int mi = 0; mi < 4; ++mi) {
            int t0 = row0 + mi * 16 + 8 * hh;    // 8-aligned
            int bb = t0 >> 11;
            int tt = t0 & 2047;
            union { bf16x8 v; __bf16 e[8]; } st;
#pragma unroll
            for (int r = 0; r < 8; ++r) st.e[r] = (__bf16)(acc[mi][r] + bv);
            *(bf16x8*)(Vo + (((size_t)(bb * Hsz + head)) * DHsz + dh) * Tsz + tt) = st.v;
        }
    } else {
        __bf16* dst = (which == 0) ? Qo : Ko;
#pragma unroll
        for (int mi = 0; mi < 4; ++mi) {
#pragma unroll
            for (int r = 0; r < 8; ++r) {
                int m = row0 + mi * 16 + r + 8 * hh;
                int bb = m >> 11;
                int t = m & 2047;
                dst[(((size_t)(bb * Hsz + head)) * Tsz + t) * DHsz + dh] =
                    (__bf16)(acc[mi][r] + bv);
            }
        }
    }
}

// ---------------------------------------------------------------------------
// Kernel 2: causal flash attention per (b, h, 64-query tile).
// Block = 128 threads (4 waves); each wave owns 16 query rows, fully
// independent (no block barriers). V is pre-transposed [DH][T] per head, so
// both K and V B-fragments are contiguous global b128 loads. P re-layout via
// per-wave LDS guarded by s_wait_dscnt.
// ---------------------------------------------------------------------------
__global__ __launch_bounds__(128) void attn_kernel(
    const __bf16* __restrict__ Q, const __bf16* __restrict__ K,
    const __bf16* __restrict__ V, __bf16* __restrict__ AO)
{
    __shared__ alignas(16) __bf16 Pb[4 * 16 * 40];   // per-wave P tile [q][key]

    const int lane = threadIdx.x & 31;
    const int wave = threadIdx.x >> 5;
    const int hh = lane >> 4, ln = lane & 15;
    const int q0 = blockIdx.x * 64;
    const int h = blockIdx.y;
    const int b = blockIdx.z;

    const size_t headoff = ((size_t)(b * Hsz + h)) * Tsz * DHsz;
    const __bf16* Qb = Q + headoff;
    const __bf16* Kb = K + headoff;                  // [T][DH]
    const __bf16* Vb = V + headoff;                  // [DH][T] (transposed)

    const int qw = q0 + wave * 16;                   // this wave's first query row
    bf16x16 aq0 = frag_a_bf16(Qb + (size_t)qw * DHsz + 0, DHsz, lane);
    bf16x16 aq1 = frag_a_bf16(Qb + (size_t)qw * DHsz + 32, DHsz, lane);

    f32x8 o[4];
#pragma unroll
    for (int nd = 0; nd < 4; ++nd) o[nd] = (f32x8){0, 0, 0, 0, 0, 0, 0, 0};
    float mrow[8], lrow[8];
#pragma unroll
    for (int j = 0; j < 8; ++j) { mrow[j] = -1e30f; lrow[j] = 0.0f; }

    const int nkt = (qw + 47) >> 5;                  // per-wave causal bound
    __bf16* Pw = Pb + wave * (16 * 40);

    for (int kt = 0; kt < nkt; ++kt) {
        const int kk = kt * 32;

        // Scores: S = Q K^T over DH=64 (two 32-wide k-steps), 2 key sub-tiles
        f32x8 s0 = (f32x8){0, 0, 0, 0, 0, 0, 0, 0};
        f32x8 s1 = (f32x8){0, 0, 0, 0, 0, 0, 0, 0};
        {
            bf16x16 bk;
            bk = frag_b_nmajor(Kb + (size_t)(kk) * DHsz + 0, DHsz, lane);
            s0 = wmma_bf16(aq0, bk, s0);
            bk = frag_b_nmajor(Kb + (size_t)(kk) * DHsz + 32, DHsz, lane);
            s0 = wmma_bf16(aq1, bk, s0);
            bk = frag_b_nmajor(Kb + (size_t)(kk + 16) * DHsz + 0, DHsz, lane);
            s1 = wmma_bf16(aq0, bk, s1);
            bk = frag_b_nmajor(Kb + (size_t)(kk + 16) * DHsz + 32, DHsz, lane);
            s1 = wmma_bf16(aq1, bk, s1);
        }

        // Online softmax (row = r + 8*hh, spread over 16 lanes of same hh group)
#pragma unroll
        for (int j = 0; j < 8; ++j) {
            int qa = qw + j + 8 * hh;
            float v0 = s0[j] * 0.125f;               // 1/sqrt(64)
            float v1 = s1[j] * 0.125f;
            if (kk + ln > qa) v0 = -1e30f;
            if (kk + 16 + ln > qa) v1 = -1e30f;
            float loc = fmaxf(v0, v1);
            loc = fmaxf(loc, __shfl_xor(loc, 1, 16));
            loc = fmaxf(loc, __shfl_xor(loc, 2, 16));
            loc = fmaxf(loc, __shfl_xor(loc, 4, 16));
            loc = fmaxf(loc, __shfl_xor(loc, 8, 16));
            float mnew = fmaxf(mrow[j], loc);
            float alpha = __expf(mrow[j] - mnew);
            float p0 = __expf(v0 - mnew);
            float p1 = __expf(v1 - mnew);
            float ps = p0 + p1;
            ps += __shfl_xor(ps, 1, 16);
            ps += __shfl_xor(ps, 2, 16);
            ps += __shfl_xor(ps, 4, 16);
            ps += __shfl_xor(ps, 8, 16);
            lrow[j] = lrow[j] * alpha + ps;
            mrow[j] = mnew;
#pragma unroll
            for (int nd = 0; nd < 4; ++nd) o[nd][j] *= alpha;
            Pw[(j + 8 * hh) * 40 + ln] = (__bf16)p0;
            Pw[(j + 8 * hh) * 40 + 16 + ln] = (__bf16)p1;
        }
        // cross-lane LDS round-trip within the wave: make stores visible
        asm volatile("s_wait_dscnt 0" ::: "memory");

        // O += P @ V  (P: 16x32 A-fragment from LDS; V: contiguous global)
        bf16x16 ap = frag_a_bf16(Pw, 40, lane);
#pragma unroll
        for (int nd = 0; nd < 4; ++nd) {
            bf16x16 bv = frag_b_nmajor(Vb + (size_t)(nd * 16) * Tsz + kk, Tsz, lane);
            o[nd] = wmma_bf16(ap, bv, o[nd]);
        }
    }

    // Epilogue: normalize by row sum, merge heads -> AO[b, t, h*64 + d] bf16
#pragma unroll
    for (int nd = 0; nd < 4; ++nd) {
#pragma unroll
        for (int j = 0; j < 8; ++j) {
            int t = qw + j + 8 * hh;
            int col = h * DHsz + nd * 16 + ln;
            AO[((size_t)(b * Tsz + t)) * Dsz + col] = (__bf16)(o[nd][j] / lrow[j]);
        }
    }
}

// ---------------------------------------------------------------------------
// Kernel 3: output projection  out[8192 x 1024] = AO @ W_proj + b_proj (f32)
// A is bf16 in global (written by kernel 2): direct b128 fragment loads.
// ---------------------------------------------------------------------------
__global__ __launch_bounds__(128) void proj_gemm_kernel(
    const __bf16* __restrict__ A, const float* __restrict__ W,
    const float* __restrict__ bias, float* __restrict__ Out)
{
    __shared__ alignas(16) __bf16 Bt[64 * 40];
    const int lane = threadIdx.x & 31;
    const int wave = threadIdx.x >> 5;
    const int hh = lane >> 4, ln = lane & 15;
    const int n0 = blockIdx.x * 64;
    const int row0 = blockIdx.y * 64;

    f32x8 acc[4];
#pragma unroll
    for (int mi = 0; mi < 4; ++mi) acc[mi] = (f32x8){0, 0, 0, 0, 0, 0, 0, 0};

    for (int k0 = 0; k0 < Dsz; k0 += 32) {
        __syncthreads();
#pragma unroll
        for (int i = 0; i < 4; ++i) {
            int qi = threadIdx.x + i * 128;
            int kk = qi >> 4;
            int nq = (qi & 15) * 4;
            float4 wv = *(const float4*)(W + (size_t)(k0 + kk) * Dsz + n0 + nq);
            Bt[(nq + 0) * 40 + kk] = (__bf16)wv.x;
            Bt[(nq + 1) * 40 + kk] = (__bf16)wv.y;
            Bt[(nq + 2) * 40 + kk] = (__bf16)wv.z;
            Bt[(nq + 3) * 40 + kk] = (__bf16)wv.w;
        }
        __syncthreads();

        bf16x16 bf = frag_b_nmajor(&Bt[wave * 16 * 40], 40, lane);
#pragma unroll
        for (int mi = 0; mi < 4; ++mi) {
            bf16x16 af = frag_a_bf16(A + (size_t)(row0 + mi * 16) * Dsz + k0, Dsz, lane);
            acc[mi] = wmma_bf16(af, bf, acc[mi]);
        }
    }

    const int nn = n0 + wave * 16 + ln;
    const float bv = bias[nn];
#pragma unroll
    for (int mi = 0; mi < 4; ++mi) {
#pragma unroll
        for (int r = 0; r < 8; ++r) {
            int m = row0 + mi * 16 + r + 8 * hh;
            Out[(size_t)m * Dsz + nn] = acc[mi][r] + bv;
        }
    }
}

// ---------------------------------------------------------------------------
// Host launcher
// inputs: x, mask(unused; causal mask computed inline), W_qkv, b_qkv, W_proj, b_proj
// ws layout: Q | K | V(transposed) | AO  (each B*H*T*DH bf16 = 16 MB, total 64 MB)
// ---------------------------------------------------------------------------
extern "C" void kernel_launch(void* const* d_in, const int* in_sizes, int n_in,
                              void* d_out, int out_size, void* d_ws, size_t ws_size,
                              hipStream_t stream) {
    (void)in_sizes; (void)n_in; (void)out_size; (void)ws_size;
    const float* x     = (const float*)d_in[0];
    const float* Wqkv  = (const float*)d_in[2];
    const float* bqkv  = (const float*)d_in[3];
    const float* Wproj = (const float*)d_in[4];
    const float* bproj = (const float*)d_in[5];
    float* out = (float*)d_out;

    const size_t per = (size_t)Bsz * Hsz * Tsz * DHsz;   // elements
    __bf16* Qw  = (__bf16*)d_ws;
    __bf16* Kw  = Qw + per;
    __bf16* Vw  = Kw + per;   // transposed layout [B,H,DH,T]
    __bf16* AOw = Vw + per;

    qkv_gemm_kernel<<<dim3(3 * Dsz / 64, (Bsz * Tsz) / 64), 128, 0, stream>>>(
        x, Wqkv, bqkv, Qw, Kw, Vw);
    attn_kernel<<<dim3(Tsz / 64, Hsz, Bsz), 128, 0, stream>>>(Qw, Kw, Vw, AOw);
    proj_gemm_kernel<<<dim3(Dsz / 64, (Bsz * Tsz) / 64), 128, 0, stream>>>(
        AOw, Wproj, bproj, out);
}
